// HieCoAtten_8778913153609
// MI455X (gfx1250) — compile-verified, matrix operand order checked by reading, not run on hardware
//
#include <hip/hip_runtime.h>
#include <hip/hip_bf16.h>
#include <math.h>

typedef __bf16 bf16_t;
typedef __attribute__((ext_vector_type(16))) __bf16 v16bf;
typedef __attribute__((ext_vector_type(8)))  __bf16 v8bf;
typedef __attribute__((ext_vector_type(8)))  float  v8f;

#define WMMA_BF16(a, b, c) \
  __builtin_amdgcn_wmma_f32_16x16x32_bf16(false, (a), false, (b), (short)0, (c), false, false)

namespace {
constexpr int kN   = 256;
constexpr int kLV  = 196;
constexpr int kLQ  = 22;
constexpr int kIMG = 1024;
constexpr int kD   = 512;
constexpr int kOUT = 3000;
constexpr int kMI  = kN * kLV;     // 50176 image rows   (divisible by 128)
constexpr int kMQ  = kN * kLQ;     // 5632 question rows (divisible by 128)
constexpr int kMIp = kMI + 16;     // pad so padded B tiles (v up to 207) stay in-bounds
constexpr int kMQp = kMQ + 32;     // pad so padded A tiles (q up to 31) stay in-bounds
constexpr int kOUTp = 3072;        // 3000 padded so N grid divides by 128
constexpr int kXK  = 2 * kD;       // 1024
}

// ---------------------------------------------------------------------------
// bf16 fragment loader: A(16x32) / B(32x16) from row-major K-contiguous data.
// lane<16 : row=lane,    K {0..7} then {16..23}
// lane>=16: row=lane-16, K {8..15} then {24..31}
// => base + r*ld + half*8, two 16B vector loads at +0 and +16.
// ---------------------------------------------------------------------------
__device__ __forceinline__ v16bf load_frag_bf16(const bf16_t* __restrict__ p0, int ld) {
  const int lane = threadIdx.x & 31;
  const int r    = lane & 15;
  const int h    = lane >> 4;
  const bf16_t* p = p0 + (size_t)r * ld + h * 8;
  v8bf lo = *reinterpret_cast<const v8bf*>(p);
  v8bf hi = *reinterpret_cast<const v8bf*>(p + 16);
  v16bf f;
#pragma unroll
  for (int i = 0; i < 8; ++i) { f[i] = lo[i]; f[i + 8] = hi[i]; }
  return f;
}

// ---------------------------------------------------------------------------
// Generic D = A(MxK) * W(NxK)^T + bias, bf16 in, fp32 accumulate.
// Block: 128 thr (4 waves) -> 128x128 tile; each wave 64x64 (4x4 WMMA tiles).
// 8 fragment loads feed 16 WMMAs per k-step (0.5 frag/WMMA operand traffic).
// EPI: 0=none, 1=relu.  Nvalid guards stores (N may be padded for the grid).
// ---------------------------------------------------------------------------
template <int EPI, bool STORE_F, bool STORE_B>
__global__ __launch_bounds__(128) void gemm_wmma(
    const bf16_t* __restrict__ A, const bf16_t* __restrict__ W,
    const float* __restrict__ bias, float* __restrict__ Of,
    bf16_t* __restrict__ Ob, int M, int Nvalid, int K, int ldc) {
  const int w  = threadIdx.x >> 5;
  const int m0 = blockIdx.y * 128 + (w >> 1) * 64;
  const int n0 = blockIdx.x * 128 + (w & 1) * 64;

  const v8f vzero = {0.f, 0.f, 0.f, 0.f, 0.f, 0.f, 0.f, 0.f};
  v8f acc[4][4];
#pragma unroll
  for (int i = 0; i < 4; ++i)
#pragma unroll
    for (int j = 0; j < 4; ++j) acc[i][j] = vzero;

  const bf16_t* ap[4];
  const bf16_t* bp[4];
#pragma unroll
  for (int t = 0; t < 4; ++t) {
    ap[t] = A + (size_t)(m0 + 16 * t) * K;
    bp[t] = W + (size_t)(n0 + 16 * t) * K;
  }

  for (int kk = 0; kk < K; kk += 32) {
    if (kk + 64 <= K) {  // L2 prefetch of next k-step (global_prefetch_b8)
      __builtin_prefetch(ap[0] + kk + 32, 0, 1);
      __builtin_prefetch(bp[0] + kk + 32, 0, 1);
    }
    v16bf a[4], b[4];
#pragma unroll
    for (int t = 0; t < 4; ++t) a[t] = load_frag_bf16(ap[t] + kk, K);
#pragma unroll
    for (int t = 0; t < 4; ++t) b[t] = load_frag_bf16(bp[t] + kk, K);
#pragma unroll
    for (int i = 0; i < 4; ++i)
#pragma unroll
      for (int j = 0; j < 4; ++j) acc[i][j] = WMMA_BF16(a[i], b[j], acc[i][j]);
  }

  const int lane = threadIdx.x & 31;
  const int lc = lane & 15, h = lane >> 4;
#pragma unroll
  for (int i = 0; i < 4; ++i)
#pragma unroll
    for (int j = 0; j < 4; ++j) {
      const int col = n0 + j * 16 + lc;
      const float bv = (col < Nvalid) ? bias[col] : 0.f;
#pragma unroll
      for (int g = 0; g < 8; ++g) {
        const int row = m0 + i * 16 + h * 8 + g;
        if (row < M && col < Nvalid) {
          float val = acc[i][j][g] + bv;
          if (EPI == 1) val = fmaxf(val, 0.f);
          if (STORE_F) Of[(size_t)row * ldc + col] = val;
          if (STORE_B) Ob[(size_t)row * ldc + col] = (bf16_t)val;
        }
      }
    }
}

// ---------------------------------------------------------------------------
// Batched C[n] = tanh(Cq[n](22x512) @ Cv[n](196x512)^T), WMMA over padded
// 32x208 tiles. One wave per 16x16 output tile (26 tiles/batch).
// ---------------------------------------------------------------------------
__global__ __launch_bounds__(128) void batched_c_wmma(
    const bf16_t* __restrict__ Cq, const bf16_t* __restrict__ Cv,
    float* __restrict__ C) {
  const int n = blockIdx.x;
  const int wave = threadIdx.x >> 5;
  const v8f vzero = {0.f, 0.f, 0.f, 0.f, 0.f, 0.f, 0.f, 0.f};
  const int lane = threadIdx.x & 31;
  const int lc = lane & 15, h = lane >> 4;

  for (int t = wave; t < 26; t += 4) {
    const int qt = t / 13;
    const int vt = t - qt * 13;
    const bf16_t* ap = Cq + ((size_t)n * kLQ + qt * 16) * kD;
    const bf16_t* bp = Cv + ((size_t)n * kLV + vt * 16) * kD;
    v8f acc = vzero;
    for (int kk = 0; kk < kD; kk += 32) {
      v16bf a = load_frag_bf16(ap + kk, kD);
      v16bf b = load_frag_bf16(bp + kk, kD);
      acc = WMMA_BF16(a, b, acc);
    }
    const int col = vt * 16 + lc;
#pragma unroll
    for (int g = 0; g < 8; ++g) {
      const int q = qt * 16 + h * 8 + g;
      if (q < kLQ && col < kLV)
        C[((size_t)n * kLQ + q) * kLV + col] = tanhf(acc[g]);
    }
  }
}

// ----------------------------- elementwise ---------------------------------
__global__ void cvt_f2b(const float* __restrict__ in, bf16_t* __restrict__ out, int n) {
  int i = blockIdx.x * 256 + threadIdx.x;
  if (i < n) out[i] = (bf16_t)in[i];
}

__global__ void cvt_pad_rows(const float* __restrict__ in, bf16_t* __restrict__ out,
                             int rows_valid, int rows_pad, int cols) {
  int i = blockIdx.x * 256 + threadIdx.x;
  if (i >= rows_pad * cols) return;
  int r = i / cols;
  out[i] = (r < rows_valid) ? (bf16_t)in[i] : (bf16_t)0.f;
}

__global__ void gather_que(const int* __restrict__ idx, const float* __restrict__ emb,
                           float* __restrict__ qf, bf16_t* __restrict__ qb) {
  int i = blockIdx.x * 256 + threadIdx.x;
  if (i >= kMQ * kD) return;
  int row = i >> 9, d = i & (kD - 1);
  float v = emb[(size_t)idx[row] * kD + d];
  qf[i] = v;
  qb[i] = (bf16_t)v;
}

// score_v[n,v] = W_hv . tanh(img_[n,v,:] + sum_q que_[n,q,:] * C[n,q,v]) + b_hv
__global__ __launch_bounds__(256) void score_v_k(
    const float* __restrict__ imgp, const float* __restrict__ quep,
    const float* __restrict__ C, const float* __restrict__ Whv,
    const float* __restrict__ bhv, float* __restrict__ sv) {
  const int n = blockIdx.x / kLV, v = blockIdx.x % kLV;
  __shared__ float cc[kLQ];
  __shared__ float buf[256];
  if (threadIdx.x < kLQ) cc[threadIdx.x] = C[((size_t)n * kLQ + threadIdx.x) * kLV + v];
  __syncthreads();
  float acc = 0.f;
  for (int d = threadIdx.x; d < kD; d += 256) {
    float t = imgp[((size_t)n * kLV + v) * kD + d];
#pragma unroll
    for (int q = 0; q < kLQ; ++q) t += quep[((size_t)n * kLQ + q) * kD + d] * cc[q];
    acc += tanhf(t) * Whv[d];
  }
  buf[threadIdx.x] = acc;
  __syncthreads();
  for (int o = 128; o > 0; o >>= 1) {
    if (threadIdx.x < o) buf[threadIdx.x] += buf[threadIdx.x + o];
    __syncthreads();
  }
  if (threadIdx.x == 0) sv[blockIdx.x] = buf[0] + bhv[0];
}

// score_q[n,q] = W_hq . tanh(que_[n,q,:] + sum_v img_[n,v,:] * C[n,q,v]) + b_hq
// img_ (102MB) is L2-resident, so the 22x reuse across q hits L2 not HBM.
__global__ __launch_bounds__(256) void score_q_k(
    const float* __restrict__ quep, const float* __restrict__ imgp,
    const float* __restrict__ C, const float* __restrict__ Whq,
    const float* __restrict__ bhq, float* __restrict__ sq) {
  const int n = blockIdx.x / kLQ, q = blockIdx.x % kLQ;
  __shared__ float cc[kLV];
  __shared__ float buf[256];
  for (int i = threadIdx.x; i < kLV; i += 256)
    cc[i] = C[((size_t)n * kLQ + q) * kLV + i];
  __syncthreads();
  float acc = 0.f;
  for (int d = threadIdx.x; d < kD; d += 256) {
    float t = quep[((size_t)n * kLQ + q) * kD + d];
    for (int v = 0; v < kLV; ++v) t += imgp[((size_t)n * kLV + v) * kD + d] * cc[v];
    acc += tanhf(t) * Whq[d];
  }
  buf[threadIdx.x] = acc;
  __syncthreads();
  for (int o = 128; o > 0; o >>= 1) {
    if (threadIdx.x < o) buf[threadIdx.x] += buf[threadIdx.x + o];
    __syncthreads();
  }
  if (threadIdx.x == 0) sq[blockIdx.x] = buf[0] + bhq[0];
}

__global__ __launch_bounds__(256) void softmax196(const float* __restrict__ s,
                                                  float* __restrict__ a_ws,
                                                  float* __restrict__ a_out) {
  const int n = blockIdx.x, t = threadIdx.x;
  __shared__ float buf[256];
  float x = (t < kLV) ? s[(size_t)n * kLV + t] : -1e30f;
  buf[t] = x;
  __syncthreads();
  for (int o = 128; o > 0; o >>= 1) {
    if (t < o) buf[t] = fmaxf(buf[t], buf[t + o]);
    __syncthreads();
  }
  const float m = buf[0];
  __syncthreads();
  float e = (t < kLV) ? __expf(x - m) : 0.f;
  buf[t] = e;
  __syncthreads();
  for (int o = 128; o > 0; o >>= 1) {
    if (t < o) buf[t] += buf[t + o];
    __syncthreads();
  }
  const float inv = 1.f / buf[0];
  if (t < kLV) {
    float a = e * inv;
    a_ws[(size_t)n * kLV + t] = a;
    a_out[(size_t)n * kLV + t] = a;
  }
}

__global__ void softmax22(const float* __restrict__ s, float* __restrict__ a_ws,
                          float* __restrict__ a_out) {
  const int n = blockIdx.x, t = threadIdx.x;
  float x = (t < kLQ) ? s[(size_t)n * kLQ + t] : -1e30f;
  float m = x;
  for (int o = 16; o > 0; o >>= 1) m = fmaxf(m, __shfl_xor(m, o, 32));
  float e = (t < kLQ) ? __expf(x - m) : 0.f;
  float sum = e;
  for (int o = 16; o > 0; o >>= 1) sum += __shfl_xor(sum, o, 32);
  if (t < kLQ) {
    float a = e / sum;
    a_ws[(size_t)n * kLQ + t] = a;
    a_out[(size_t)n * kLQ + t] = a;
  }
}

__global__ __launch_bounds__(256) void weighted_v(const float* __restrict__ av,
                                                  const float* __restrict__ img,
                                                  float* __restrict__ xflat) {
  const int n = blockIdx.x;
  for (int d = threadIdx.x; d < kD; d += 256) {
    float acc = 0.f;
    for (int v = 0; v < kLV; ++v)
      acc += av[(size_t)n * kLV + v] * img[((size_t)n * kLV + v) * kD + d];
    xflat[(size_t)n * kD + d] = acc;
  }
}

__global__ __launch_bounds__(256) void weighted_q(const float* __restrict__ aq,
                                                  const float* __restrict__ que,
                                                  float* __restrict__ xflat) {
  const int n = blockIdx.x;
  for (int d = threadIdx.x; d < kD; d += 256) {
    float acc = 0.f;
    for (int q = 0; q < kLQ; ++q)
      acc += aq[(size_t)n * kLQ + q] * que[((size_t)n * kLQ + q) * kD + d];
    xflat[(size_t)n * kD + d] = acc;
  }
}

// ---------------------------------------------------------------------------
extern "C" void kernel_launch(void* const* d_in, const int* in_sizes, int n_in,
                              void* d_out, int out_size, void* d_ws, size_t ws_size,
                              hipStream_t stream) {
  (void)in_sizes; (void)n_in; (void)out_size; (void)ws_size;

  const float* img_feat = (const float*)d_in[0];
  const int*   que_idx  = (const int*)d_in[1];
  const float* W_img = (const float*)d_in[2];  const float* b_img = (const float*)d_in[3];
  const float* emb   = (const float*)d_in[4];
  const float* W_bv  = (const float*)d_in[5];  const float* b_bv  = (const float*)d_in[6];
  const float* W_v   = (const float*)d_in[7];  const float* b_v   = (const float*)d_in[8];
  const float* W_q   = (const float*)d_in[9];  const float* b_q   = (const float*)d_in[10];
  const float* W_hv  = (const float*)d_in[11]; const float* b_hv  = (const float*)d_in[12];
  const float* W_hq  = (const float*)d_in[13]; const float* b_hq  = (const float*)d_in[14];
  const float* W_fc  = (const float*)d_in[15]; const float* b_fc  = (const float*)d_in[16];

  // Workspace carve-up (256B aligned slices).
  char* wsp = (char*)d_ws;
  size_t off = 0;
  auto take = [&](size_t bytes) -> char* {
    char* p = wsp + off;
    off += (bytes + 255) & ~(size_t)255;
    return p;
  };
  bf16_t* imgfeat_b = (bf16_t*)take((size_t)kMI * kIMG * 2);
  bf16_t* Wimg_b = (bf16_t*)take((size_t)kD * kIMG * 2);
  bf16_t* Wbv_b  = (bf16_t*)take((size_t)kD * kD * 2);
  bf16_t* Wv_b   = (bf16_t*)take((size_t)kD * kD * 2);
  bf16_t* Wq_b   = (bf16_t*)take((size_t)kD * kD * 2);
  bf16_t* Wfc_b  = (bf16_t*)take((size_t)kOUTp * kXK * 2);
  float*  img_f  = (float*)take((size_t)kMI * kD * 4);
  bf16_t* img_b  = (bf16_t*)take((size_t)kMI * kD * 2);
  bf16_t* Cv_b   = (bf16_t*)take((size_t)kMIp * kD * 2);
  float*  imgp_f = (float*)take((size_t)kMI * kD * 4);
  float*  que_f  = (float*)take((size_t)kMQ * kD * 4);
  bf16_t* que_b  = (bf16_t*)take((size_t)kMQ * kD * 2);
  bf16_t* Cq_b   = (bf16_t*)take((size_t)kMQp * kD * 2);
  float*  quep_f = (float*)take((size_t)kMQ * kD * 4);
  float*  Cmat   = (float*)take((size_t)kN * kLQ * kLV * 4);
  float*  sv     = (float*)take((size_t)kMI * 4);
  float*  sq     = (float*)take((size_t)kMQ * 4);
  float*  av     = (float*)take((size_t)kMI * 4);
  float*  aq     = (float*)take((size_t)kMQ * 4);
  float*  xflat  = (float*)take((size_t)kN * kXK * 4);
  bf16_t* x_b    = (bf16_t*)take((size_t)kN * kXK * 2);

  float* out_x  = (float*)d_out;                 // (256,3000)
  float* out_av = out_x + (size_t)kN * kOUT;     // (256,196)
  float* out_aq = out_av + (size_t)kN * kLV;     // (256,22)

  auto nb = [](size_t n) { return (unsigned)((n + 255) / 256); };

  // 1) bf16 conversions of activations/weights
  cvt_f2b<<<nb((size_t)kMI * kIMG), 256, 0, stream>>>(img_feat, imgfeat_b, kMI * kIMG);
  cvt_f2b<<<nb((size_t)kD * kIMG), 256, 0, stream>>>(W_img, Wimg_b, kD * kIMG);
  cvt_f2b<<<nb((size_t)kD * kD), 256, 0, stream>>>(W_bv, Wbv_b, kD * kD);
  cvt_f2b<<<nb((size_t)kD * kD), 256, 0, stream>>>(W_v, Wv_b, kD * kD);
  cvt_f2b<<<nb((size_t)kD * kD), 256, 0, stream>>>(W_q, Wq_b, kD * kD);
  cvt_pad_rows<<<nb((size_t)kOUTp * kXK), 256, 0, stream>>>(W_fc, Wfc_b, kOUT, kOUTp, kXK);

  // 2) img = relu(img_feat @ W_img^T + b_img)   [fp32 + bf16 copies]
  gemm_wmma<1, true, true><<<dim3(kD / 128, kMI / 128), 128, 0, stream>>>(
      imgfeat_b, Wimg_b, b_img, img_f, img_b, kMI, kD, kIMG, kD);

  // 3) Cv = img @ W_bv^T + b_bv  (bf16, feeds batched C)
  gemm_wmma<0, false, true><<<dim3(kD / 128, kMI / 128), 128, 0, stream>>>(
      img_b, Wbv_b, b_bv, nullptr, Cv_b, kMI, kD, kD, kD);
  // 4) img_ = img @ W_v^T + b_v  (fp32)
  gemm_wmma<0, true, false><<<dim3(kD / 128, kMI / 128), 128, 0, stream>>>(
      img_b, Wv_b, b_v, imgp_f, nullptr, kMI, kD, kD, kD);

  // 5) que = emb[que_features]
  gather_que<<<nb((size_t)kMQ * kD), 256, 0, stream>>>(que_idx, emb, que_f, que_b);

  // 6) Cq = que @ W_bv^T + b_bv ; que_ = que @ W_q^T + b_q
  gemm_wmma<0, false, true><<<dim3(kD / 128, kMQ / 128), 128, 0, stream>>>(
      que_b, Wbv_b, b_bv, nullptr, Cq_b, kMQ, kD, kD, kD);
  gemm_wmma<0, true, false><<<dim3(kD / 128, kMQ / 128), 128, 0, stream>>>(
      que_b, Wq_b, b_q, quep_f, nullptr, kMQ, kD, kD, kD);

  // 7) C = tanh(batched Cq @ Cv^T)
  batched_c_wmma<<<kN, 128, 0, stream>>>(Cq_b, Cv_b, Cmat);

  // 8) attention scores (fused, Hv/Hq never materialized)
  score_v_k<<<kMI, 256, 0, stream>>>(imgp_f, quep_f, Cmat, W_hv, b_hv, sv);
  score_q_k<<<kMQ, 256, 0, stream>>>(quep_f, imgp_f, Cmat, W_hq, b_hq, sq);

  // 9) softmaxes (write outputs av/aq too)
  softmax196<<<kN, 256, 0, stream>>>(sv, av, out_av);
  softmax22<<<kN, 32, 0, stream>>>(sq, aq, out_aq);

  // 10) v/q pooled features -> flat x buffer ([v.flat, q.flat] == concat/reshape)
  weighted_v<<<kN, 256, 0, stream>>>(av, img_f, xflat);
  weighted_q<<<kN, 256, 0, stream>>>(aq, que_f, xflat + (size_t)kN * kD);
  cvt_f2b<<<nb((size_t)kN * kXK), 256, 0, stream>>>(xflat, x_b, kN * kXK);

  // 11) x @ W_fc^T + b_fc -> out (N padded 3000->3072, store-guarded)
  gemm_wmma<0, true, false><<<dim3(kOUTp / 128, kN / 128), 128, 0, stream>>>(
      x_b, Wfc_b, b_fc, out_x, nullptr, kN, kOUT, kXK, kOUT);
}
